// G_5643587027142
// MI455X (gfx1250) — matrix operand order repaired and checked by reading, not compiled, not run on hardware
//
#include <hip/hip_runtime.h>
#include <hip/hip_bf16.h>
#include <stdint.h>

typedef __attribute__((ext_vector_type(16))) _Float16 v16h;
typedef __attribute__((ext_vector_type(8)))  float    v8f;
typedef __attribute__((ext_vector_type(4)))  unsigned int u32x4;
typedef __attribute__((ext_vector_type(8)))  unsigned int u32x8;

union FragH { v16h v; uint4 q[2]; };

__constant__ float POSC[31][3] = {
  {-11.16231f, 1.71701f, 9.612973e-06f},
  {-12.52701f, -0.008990765f, -2.464731e-08f},
  {-11.69302f, -0.004831314f, -1.340959e-08f},
  {-7.388115f, -2.221376f, 6.07915e-06f},
  {-5.895767f, 0.09526443f, 0.0003307685f},
  {4.509056f, 1.610937f, 5.190379f},
  {-5.375591f, 0.0f, 7.629395e-06f},
  {-18.61573f, -9.536743e-07f, -7.629395e-06f},
  {-18.90229f, 0.0f, 0.0f},
  {-1.719231f, 0.5651627f, -1.910677f},
  {-5.540304f, -0.2278214f, -1.871119f},
  {-5.579481f, -0.4572525f, -0.4858208f},
  {-5.428627f, -0.4348068f, 0.7903614f},
  {-5.161129f, -0.2240829f, 1.925165f},
  {4.509048f, 1.610966f, -5.190396f},
  {-5.37559f, 0.0f, 0.0f},
  {-18.61573f, 0.0f, 0.0f},
  {-18.9023f, -1.907349e-06f, 0.0f},
  {-1.719234f, 0.5651703f, 1.910675f},
  {-5.540302f, -0.2278137f, 1.871119f},
  {-5.579479f, -0.4572525f, 0.4858189f},
  {-5.428623f, -0.4348068f, -0.7903633f},
  {-5.161131f, -0.2240753f, -1.925165f},
  {11.23972f, -1.101868f, 6.513383f},
  {-32.66669f, 1.192093e-07f, 0.0f},
  {-36.31458f, -9.536743e-07f, 0.0f},
  {-8.059542f, 8.277123f, 0.0f},
  {11.23971f, -1.101831f, -6.513384f},
  {-32.66669f, 0.0f, 0.0f},
  {-36.31458f, -4.768372e-07f, -4.768372e-07f},
  {-8.059542f, 8.277124f, -4.768372e-07f}
};

// ------------------------------------------------------------------
// Weight / input packing (fp32 -> f16, padded layouts for WMMA)
// ------------------------------------------------------------------
__global__ void prep_w1_k(const float* __restrict__ w1, _Float16* __restrict__ A) {
  int i = blockIdx.x * blockDim.x + threadIdx.x;
  if (i >= 4096 * 128) return;
  int m = i >> 7, k = i & 127;
  A[i] = (k < 100) ? (_Float16)w1[(size_t)k * 4096 + m] : (_Float16)0.0f;
}

__global__ void prep_x_k(const float* __restrict__ x, _Float16* __restrict__ B) {
  int i = blockIdx.x * blockDim.x + threadIdx.x;
  if (i >= 512 * 128) return;
  int n = i >> 7, k = i & 127;
  B[i] = (k < 100) ? (_Float16)x[n * 100 + k] : (_Float16)0.0f;
}

// convT weights: w [IC][OC][4][4] fp32 -> wt [kk=ky*4+kx][OCpad][IC] f16
__global__ void prep_wt_k(const float* __restrict__ w, _Float16* __restrict__ wt,
                          int IC, int OC, int OCpad, int total) {
  int i = blockIdx.x * blockDim.x + threadIdx.x;
  if (i >= total) return;
  int ic  = i % IC;
  int rem = i / IC;
  int oc  = rem % OCpad;
  int kk  = rem / OCpad;
  wt[i] = (oc < OC) ? (_Float16)w[((size_t)ic * OC + oc) * 16 + kk] : (_Float16)0.0f;
}

// ------------------------------------------------------------------
// Layer 1: C[512][4096] = X[512][128] * W[4096][128]^T  (WMMA f16->f32)
// ------------------------------------------------------------------
__global__ __launch_bounds__(256) void gemm_l1_k(const _Float16* __restrict__ A,
                                                 const _Float16* __restrict__ B,
                                                 float* __restrict__ out) {
  int wave = blockIdx.x * (blockDim.x >> 5) + (threadIdx.x >> 5);
  if (wave >= 256 * 32) return;
  int tn = wave / 256, tm = wave % 256;
  int l  = threadIdx.x & 31;
  int lh = l >> 4, ll = l & 15;

  const _Float16* ap = A + (size_t)(tm * 16 + ll) * 128 + lh * 8;
  const _Float16* bp = B + (size_t)(tn * 16 + ll) * 128 + lh * 8;

  v8f acc = {0.f, 0.f, 0.f, 0.f, 0.f, 0.f, 0.f, 0.f};
  for (int k0 = 0; k0 < 128; k0 += 32) {
    FragH fa, fb;
    fa.q[0] = *(const uint4*)(ap + k0);
    fa.q[1] = *(const uint4*)(ap + k0 + 16);
    fb.q[0] = *(const uint4*)(bp + k0);
    fb.q[1] = *(const uint4*)(bp + k0 + 16);
    acc = __builtin_amdgcn_wmma_f32_16x16x32_f16(false, fa.v, false, fb.v,
                                                 (short)0, acc, false, false);
  }
  size_t o = (size_t)(tn * 16 + ll) * 4096 + tm * 16 + 8 * lh;
  *(float4*)(out + o)     = make_float4(acc[0], acc[1], acc[2], acc[3]);
  *(float4*)(out + o + 4) = make_float4(acc[4], acc[5], acc[6], acc[7]);
}

// InstanceNorm+ReLU for layer1 (spatial 2x2) -> NHWC f16
__global__ void inorm_l1_k(const float* __restrict__ c, _Float16* __restrict__ a1) {
  int t = blockIdx.x * blockDim.x + threadIdx.x;
  if (t >= 512 * 1024) return;
  int b = t >> 10, oc = t & 1023;
  const float* p = c + (size_t)b * 4096 + oc * 4;
  float v0 = p[0], v1 = p[1], v2 = p[2], v3 = p[3];
  float m  = (v0 + v1 + v2 + v3) * 0.25f;
  float sq = (v0 * v0 + v1 * v1 + v2 * v2 + v3 * v3) * 0.25f;
  float rs = rsqrtf(sq - m * m + 1e-5f);
  float vv[4] = {v0, v1, v2, v3};
#pragma unroll
  for (int ppos = 0; ppos < 4; ++ppos) {
    float val = fmaxf((vv[ppos] - m) * rs, 0.0f);
    int y = ppos >> 1, xx = ppos & 1;
    a1[((size_t)(b * 2 + y) * 2 + xx) * 1024 + oc] = (_Float16)val;
  }
}

// ------------------------------------------------------------------
// ConvTranspose k4/s2/p1 as block-tiled implicit WMMA GEMM.
// Block = 256 thr = 8 waves (2 M x 4 N), tile 64(OC) x 64(pixels of one
// parity class). A (weights) staged via Tensor Data Mover, B (activations)
// staged via async global->LDS b128 loads (boundary pixels zeroed via DS).
// ------------------------------------------------------------------
__global__ __launch_bounds__(256) void convt_wmma_k(const _Float16* __restrict__ act,
                                                    const _Float16* __restrict__ wt,
                                                    float* __restrict__ out,
                                                    int IC, int OCpad, int OCstore,
                                                    int IH, int IW,
                                                    int tilesM, int tilesN) {
  __shared__ __align__(16) _Float16 ldsA[64 * 128];   // 16 KB weight tile
  __shared__ __align__(16) _Float16 ldsB[64 * 128];   // 16 KB activation tile

  long long per_cls = (long long)tilesM * tilesN;
  long long blk = blockIdx.x;
  if (blk >= 4 * per_cls) return;
  int cls = (int)(blk / per_cls);
  int rem = (int)(blk % per_cls);
  int tnB = rem / tilesM, tmB = rem % tilesM;
  int ry = cls >> 1, rx = cls & 1;

  int tid = threadIdx.x;
  int wv  = tid >> 5;
  int mw  = wv >> 2;            // 0..1 : M group of this wave
  int nw  = wv & 3;             // 0..3 : N group of this wave
  int l   = tid & 31;
  int lh  = l >> 4, ll = l & 15;

  int SP = IH * IW;
  int n  = tnB * 64 + nw * 16 + ll;        // this lane's output pixel
  int b  = n / SP, sp = n % SP;
  int my = sp / IW, mx = sp % IW;

  v8f acc0 = {0.f, 0.f, 0.f, 0.f, 0.f, 0.f, 0.f, 0.f};
  v8f acc1 = {0.f, 0.f, 0.f, 0.f, 0.f, 0.f, 0.f, 0.f};

  int chunk    = (IC < 128) ? IC : 128;    // K chunk (halves)
  int segPerPx = chunk >> 3;               // 16B segments per pixel row
  int nseg     = 64 * segPerPx;

#pragma unroll
  for (int t = 0; t < 4; ++t) {
    int ty = t >> 1, tx = t & 1;
    int ky = (ry == 0) ? (ty ? 3 : 1) : (ty ? 2 : 0);
    int dy = (ry == 0) ? (ty ? -1 : 0) : (ty ? 0 : 1);
    int kx = (rx == 0) ? (tx ? 3 : 1) : (tx ? 2 : 0);
    int dx = (rx == 0) ? (tx ? -1 : 0) : (tx ? 0 : 1);
    int kkidx = ky * 4 + kx;

    for (int k0 = 0; k0 < IC; k0 += 128) {
      __syncthreads();   // previous chunk fully consumed before LDS overwrite

      // ---- A tile (64 x chunk) via TDM: one descriptor, wave 0 issues ----
      if (wv == 0) {
        unsigned long long ga = (unsigned long long)(uintptr_t)
            (wt + ((size_t)kkidx * OCpad + tmB * 64) * IC + k0);
        unsigned int lb = (unsigned int)(uintptr_t)(&ldsA[0]);
        u32x4 g0;
        u32x8 g1;
        g0[0] = 1u;                                   // count=1, user mode
        g0[1] = lb;                                   // lds_addr (bytes)
        g0[2] = (unsigned int)ga;                     // global_addr[31:0]
        g0[3] = ((unsigned int)(ga >> 32) & 0x01FFFFFFu) | 0x80000000u; // [56:32] | type=2
        unsigned int dim0  = (unsigned int)((IC - k0) >> 1);  // dwords left in row
        unsigned int tile0 = (unsigned int)(chunk >> 1);      // dwords per row
        unsigned int str0  = (unsigned int)(IC >> 1);         // row stride (dwords)
        g1[0] = 0x20000u;                             // data_size=2 (4B), no mask
        g1[1] = (dim0 & 0xFFFFu) << 16;               // tensor_dim0[15:0]
        g1[2] = (dim0 >> 16) | (64u << 16);           // tensor_dim0[31:16] | tensor_dim1
        g1[3] = (tile0 & 0xFFFFu) << 16;              // tile_dim0
        g1[4] = 64u;                                  // tile_dim1 = 64 rows
        g1[5] = str0;                                 // tensor_dim0_stride[31:0]
        g1[6] = 0u;
        g1[7] = 0u;
        asm volatile("tensor_load_to_lds %0, %1" :: "s"(g0), "s"(g1) : "memory");
        __builtin_amdgcn_s_wait_tensorcnt(0);
      }

      // ---- B tile (64 px x chunk) via async global->LDS ----
      for (int e = tid; e < nseg; e += 256) {
        int px  = e / segPerPx;
        int seg = e - px * segPerPx;
        int np  = tnB * 64 + px;
        int bb  = np / SP, spp = np % SP;
        int iy  = spp / IW + dy, ix = spp % IW + dx;
        _Float16* lp = &ldsB[px * chunk + seg * 8];
        if (iy >= 0 && iy < IH && ix >= 0 && ix < IW) {
          const _Float16* gp = act + ((size_t)((bb * IH + iy) * IW + ix)) * IC + k0 + seg * 8;
          unsigned int loff = (unsigned int)(uintptr_t)lp;
          asm volatile("global_load_async_to_lds_b128 %0, %1, off"
                       :: "v"(loff), "v"(gp) : "memory");
        } else {
          *(uint4*)lp = make_uint4(0u, 0u, 0u, 0u);   // zero padding via DS store
        }
      }
      asm volatile("s_wait_asynccnt 0x0" ::: "memory");
      __syncthreads();

      // ---- compute: two 16x16 WMMAs per K-step, shared B fragment ----
      for (int kk = 0; kk < chunk; kk += 32) {
        FragH fb, fa0, fa1;
        const _Float16* bpp = &ldsB[(nw * 16 + ll) * chunk + kk + lh * 8];
        fb.q[0] = *(const uint4*)(bpp);
        fb.q[1] = *(const uint4*)(bpp + 8 + 8);
        const _Float16* ap0 = &ldsA[(mw * 32 + ll) * chunk + kk + lh * 8];
        fa0.q[0] = *(const uint4*)(ap0);
        fa0.q[1] = *(const uint4*)(ap0 + 16);
        const _Float16* ap1 = ap0 + 16 * chunk;
        fa1.q[0] = *(const uint4*)(ap1);
        fa1.q[1] = *(const uint4*)(ap1 + 16);
        acc0 = __builtin_amdgcn_wmma_f32_16x16x32_f16(false, fa0.v, false, fb.v,
                                                      (short)0, acc0, false, false);
        acc1 = __builtin_amdgcn_wmma_f32_16x16x32_f16(false, fa1.v, false, fb.v,
                                                      (short)0, acc1, false, false);
      }
    }
  }

  // ---- store C tiles (NHWC fp32) ----
  int oy = 2 * my + ry, ox = 2 * mx + rx;
  int OH = 2 * IH, OW = 2 * IW;
  size_t pixoff = ((size_t)(b * OH + oy) * OW + ox) * OCstore;
#pragma unroll
  for (int sub = 0; sub < 2; ++sub) {
    v8f a = sub ? acc1 : acc0;
    int mbase = tmB * 64 + mw * 32 + sub * 16 + 8 * lh;
    if (mbase + 8 <= OCstore) {
      *(float4*)(out + pixoff + mbase)     = make_float4(a[0], a[1], a[2], a[3]);
      *(float4*)(out + pixoff + mbase + 4) = make_float4(a[4], a[5], a[6], a[7]);
    } else {
#pragma unroll
      for (int r = 0; r < 8; ++r)
        if (mbase + r < OCstore) out[pixoff + mbase + r] = a[r];
    }
  }
}

// ------------------------------------------------------------------
// InstanceNorm stats + apply (NHWC fp32 -> NHWC f16 with ReLU)
// ------------------------------------------------------------------
__global__ void stats_k(const float* __restrict__ o, float* __restrict__ mean,
                        float* __restrict__ rstd, int BOC, int S, int OC) {
  int t = blockIdx.x * blockDim.x + threadIdx.x;
  if (t >= BOC) return;
  int b = t / OC, oc = t % OC;
  const float* p = o + (size_t)b * S * OC + oc;
  float s = 0.f, sq = 0.f;
  for (int i = 0; i < S; ++i) {
    float v = p[(size_t)i * OC];
    s += v; sq += v * v;
  }
  float m = s / (float)S;
  mean[t] = m;
  rstd[t] = rsqrtf(sq / (float)S - m * m + 1e-5f);
}

__global__ void norm_k(const float* __restrict__ o, const float* __restrict__ mean,
                       const float* __restrict__ rstd, _Float16* __restrict__ a,
                       long long total, int S, int OC) {
  long long e = (long long)blockIdx.x * blockDim.x + threadIdx.x;
  if (e >= total) return;
  int oc = (int)(e % OC);
  long long pix = e / OC;
  int b = (int)(pix / S);
  int idx = b * OC + oc;
  float v = (o[e] - mean[idx]) * rstd[idx];
  a[e] = (_Float16)fmaxf(v, 0.0f);
}

__global__ void tanh_k(const float* __restrict__ o, const float* __restrict__ b6,
                       float* __restrict__ y, int total) {
  int e = blockIdx.x * blockDim.x + threadIdx.x;
  if (e >= total) return;
  y[e] = tanhf(o[e] + b6[0]);
}

// ------------------------------------------------------------------
// Forward kinematics (get_pos): one thread per row of y[32768][64]
// ------------------------------------------------------------------
__device__ __forceinline__ void getm_t(const float* v, float* M) {
  float t   = sqrtf(v[0] * v[0] + v[1] * v[1] + v[2] * v[2]);
  float inv = 1.0f / t;
  float rx = v[0] * inv, ry = v[1] * inv, rz = v[2] * inv;
  float c = cosf(t), s = sinf(t), o = 1.0f - c;
  float A0 = c + o * rx * rx,  A1 = o * rx * ry - s * rz, A2 = o * rx * rz + s * ry;
  float A3 = o * ry * rx + s * rz, A4 = c + o * ry * ry,  A5 = o * ry * rz - s * rx;
  float A6 = o * rz * rx - s * ry, A7 = o * rz * ry + s * rx, A8 = c + o * rz * rz;
  M[0] = A0; M[1] = A3; M[2] = A6;
  M[3] = A1; M[4] = A4; M[5] = A7;
  M[6] = A2; M[7] = A5; M[8] = A8;
}
__device__ __forceinline__ void m3mul(const float* A, const float* B, float* C) {
#pragma unroll
  for (int i = 0; i < 3; ++i)
#pragma unroll
    for (int j = 0; j < 3; ++j)
      C[i * 3 + j] = A[i * 3] * B[j] + A[i * 3 + 1] * B[3 + j] + A[i * 3 + 2] * B[6 + j];
}
__device__ __forceinline__ void step_p(const float* M, const float* v, const float* P, float* o) {
#pragma unroll
  for (int i = 0; i < 3; ++i)
    o[i] = M[i * 3] * v[0] + M[i * 3 + 1] * v[1] + M[i * 3 + 2] * v[2] + P[i];
}
__device__ __forceinline__ void cp9(float* d, const float* s) {
#pragma unroll
  for (int i = 0; i < 9; ++i) d[i] = s[i];
}
__device__ __forceinline__ void cp3(float* d, const float* s) {
  d[0] = s[0]; d[1] = s[1]; d[2] = s[2];
}

__global__ void fk_k(const float* __restrict__ y, float* __restrict__ out, int rows) {
  int r = blockIdx.x * blockDim.x + threadIdx.x;
  if (r >= rows) return;
  const float* yr = y + (size_t)r * 64;
  float* o = out + (size_t)r * 93;

  float M0[9]; getm_t(yr + 30, M0);
  float P0[3] = {POSC[0][0], POSC[0][1], POSC[0][2]};
  o[0] = P0[0]; o[1] = P0[1]; o[2] = P0[2];

  float M[9], P[3], G[9], T[9], Pn[3];
  cp9(M, M0); cp3(P, P0);
  int cot = 1;

  const int ch1[3] = {9, 8, 7};
  for (int s = 0; s < 3; ++s) {
    step_p(M, POSC[cot], P, Pn); cp3(P, Pn);
    getm_t(yr + 3 * ch1[s], G); m3mul(M, G, T); cp9(M, T);
    cp3(o + cot * 3, P); ++cot;
  }
  step_p(M, POSC[cot], P, Pn); cp3(o + cot * 3, Pn); ++cot;

  float M2[9], P2[3]; cp9(M2, M); cp3(P2, P);
  const int ch2[4] = {6, 5, 4, 3};
  for (int s = 0; s < 4; ++s) {
    step_p(M2, POSC[cot], P2, Pn); cp3(P2, Pn);
    getm_t(yr + 3 * ch2[s], G); m3mul(M2, G, T); cp9(M2, T);
    cp3(o + cot * 3, P2); ++cot;
  }
  for (int s = 0; s < 5; ++s) {
    step_p(M2, POSC[cot], P2, Pn); cp3(o + cot * 3, Pn); ++cot;
  }

  float M3[9], P3[3]; cp9(M3, M); cp3(P3, P);
  const int ch3[4] = {11, 12, 13, 14};
  for (int s = 0; s < 4; ++s) {
    step_p(M3, POSC[cot], P3, Pn); cp3(P3, Pn);
    getm_t(yr + 3 * ch3[s], G); m3mul(M3, G, T); cp9(M3, T);
    cp3(o + cot * 3, P3); ++cot;
  }
  for (int s = 0; s < 5; ++s) {
    step_p(M3, POSC[cot], P3, Pn); cp3(o + cot * 3, Pn); ++cot;
  }

  float M4[9], P4[3]; cp9(M4, M0); cp3(P4, P0);
  const int ch4[3] = {2, 1, 0};
  for (int s = 0; s < 3; ++s) {
    step_p(M4, POSC[cot], P4, Pn); cp3(P4, Pn);
    getm_t(yr + 3 * ch4[s], G); m3mul(M4, G, T); cp9(M4, T);
    cp3(o + cot * 3, P4); ++cot;
  }
  step_p(M4, POSC[cot], P4, Pn); cp3(o + cot * 3, Pn); ++cot;

  float M5[9], P5[3]; cp9(M5, M0); cp3(P5, P0);
  const int ch5[3] = {15, 16, 17};
  for (int s = 0; s < 3; ++s) {
    step_p(M5, POSC[cot], P5, Pn); cp3(P5, Pn);
    getm_t(yr + 3 * ch5[s], G); m3mul(M5, G, T); cp9(M5, T);
    cp3(o + cot * 3, P5); ++cot;
  }
  step_p(M5, POSC[cot], P5, Pn); cp3(o + cot * 3, Pn);
}

// ------------------------------------------------------------------
extern "C" void kernel_launch(void* const* d_in, const int* in_sizes, int n_in,
                              void* d_out, int out_size, void* d_ws, size_t ws_size,
                              hipStream_t stream) {
  const float* x  = (const float*)d_in[0];
  const float* w1 = (const float*)d_in[1];
  const float* w2 = (const float*)d_in[3];
  const float* w3 = (const float*)d_in[5];
  const float* w4 = (const float*)d_in[7];
  const float* w5 = (const float*)d_in[9];
  const float* w6 = (const float*)d_in[11];
  const float* b6 = (const float*)d_in[12];
  // b1..b5 cancel under InstanceNorm (mean subtraction) and are unused.

  char* ws = (char*)d_ws;
  size_t off = 0;
  auto AL = [&](size_t bytes) { size_t r = off; off += (bytes + 255) & ~(size_t)255; return r; };

  _Float16* wA1  = (_Float16*)(ws + AL((size_t)4096 * 128 * 2));
  _Float16* xB1  = (_Float16*)(ws + AL((size_t)512 * 128 * 2));
  _Float16* wt2  = (_Float16*)(ws + AL((size_t)16 * 512 * 1024 * 2));
  _Float16* wt3  = (_Float16*)(ws + AL((size_t)16 * 256 * 512 * 2));
  _Float16* wt4  = (_Float16*)(ws + AL((size_t)16 * 128 * 256 * 2));
  _Float16* wt5  = (_Float16*)(ws + AL((size_t)16 * 64 * 128 * 2));
  _Float16* wt6  = (_Float16*)(ws + AL((size_t)16 * 64 * 64 * 2));   // OC padded to 64
  _Float16* actA = (_Float16*)(ws + AL((size_t)64 << 20));
  _Float16* actB = (_Float16*)(ws + AL((size_t)32 << 20));
  float*    o32  = (float*)   (ws + AL((size_t)128 << 20));
  float*    meanB= (float*)   (ws + AL((size_t)2 << 20));
  float*    rstdB= (float*)   (ws + AL((size_t)2 << 20));
  float*    ybuf = (float*)   (ws + AL((size_t)8 << 20));

  auto cdiv = [](long long a, long long b) { return (int)((a + b - 1) / b); };

  // ---- weight / input packing ----
  prep_w1_k<<<cdiv(4096 * 128, 256), 256, 0, stream>>>(w1, wA1);
  prep_x_k <<<cdiv(512 * 128, 256),  256, 0, stream>>>(x, xB1);
  prep_wt_k<<<cdiv((long long)16 * 512 * 1024, 256), 256, 0, stream>>>(w2, wt2, 1024, 512, 512, 16 * 512 * 1024);
  prep_wt_k<<<cdiv((long long)16 * 256 * 512, 256),  256, 0, stream>>>(w3, wt3, 512, 256, 256, 16 * 256 * 512);
  prep_wt_k<<<cdiv((long long)16 * 128 * 256, 256),  256, 0, stream>>>(w4, wt4, 256, 128, 128, 16 * 128 * 256);
  prep_wt_k<<<cdiv((long long)16 * 64 * 128, 256),   256, 0, stream>>>(w5, wt5, 128, 64, 64, 16 * 64 * 128);
  prep_wt_k<<<cdiv((long long)16 * 64 * 64, 256),    256, 0, stream>>>(w6, wt6, 64, 1, 64, 16 * 64 * 64);

  // ---- layer 1 ----
  gemm_l1_k<<<cdiv((long long)256 * 32, 8), 256, 0, stream>>>(wA1, xB1, o32);
  inorm_l1_k<<<cdiv(512 * 1024, 256), 256, 0, stream>>>(o32, actA);

  auto conv_layer = [&](const _Float16* ain, const _Float16* wt, int IC, int OCpad,
                        int OCstore, int IH, int IW) {
    int tilesM = OCpad / 64;
    int tilesN = (512 * IH * IW) / 64;
    long long blocks = 4LL * tilesM * tilesN;
    convt_wmma_k<<<(int)blocks, 256, 0, stream>>>(ain, wt, o32, IC, OCpad, OCstore,
                                                  IH, IW, tilesM, tilesN);
  };
  auto norm_layer = [&](_Float16* aout, int OC, int OH, int OW) {
    int S = OH * OW;
    int BOC = 512 * OC;
    stats_k<<<cdiv(BOC, 256), 256, 0, stream>>>(o32, meanB, rstdB, BOC, S, OC);
    long long total = (long long)BOC * S;
    norm_k<<<cdiv(total, 256), 256, 0, stream>>>(o32, meanB, rstdB, aout, total, S, OC);
  };

  // ---- layers 2..5 ----
  conv_layer(actA, wt2, 1024, 512, 512, 2, 2);   norm_layer(actB, 512, 4, 4);
  conv_layer(actB, wt3,  512, 256, 256, 4, 4);   norm_layer(actA, 256, 8, 8);
  conv_layer(actA, wt4,  256, 128, 128, 8, 8);   norm_layer(actB, 128, 16, 16);
  conv_layer(actB, wt5,  128,  64,  64, 16, 16); norm_layer(actA, 64, 32, 32);

  // ---- layer 6 + tanh ----
  conv_layer(actA, wt6, 64, 64, 1, 32, 32);
  tanh_k<<<cdiv(512 * 64 * 64, 256), 256, 0, stream>>>(o32, b6, ybuf, 512 * 64 * 64);

  // ---- forward kinematics ----
  fk_k<<<cdiv(32768, 128), 128, 0, stream>>>(ybuf, (float*)d_out, 32768);
}
